// QuantGemma3Attention_13683765805112
// MI455X (gfx1250) — matrix operand order repaired and checked by reading, not compiled, not run on hardware
//
#include <hip/hip_runtime.h>
#include <hip/hip_bf16.h>
#include <cstdint>
#include <cstddef>

#define B_      2
#define S_      2048
#define H_      2560
#define NH      8
#define NKV     4
#define DH      256
#define SCALE   0.0625f      /* 256^-0.5 */
#define SOFTCAP 50.0f
#define WINDOW  1024
#define KC      64           /* K-chunk staged in LDS per GEMM iteration */

typedef __attribute__((ext_vector_type(16))) __bf16 v16bf;
typedef __attribute__((ext_vector_type(8)))  __bf16 v8bf;
typedef __attribute__((ext_vector_type(8)))  float  v8f;

__device__ __forceinline__ v8f vzero8() {
  v8f z;
#pragma unroll
  for (int i = 0; i < 8; ++i) z[i] = 0.0f;
  return z;
}

__device__ __forceinline__ v16bf pack16(const __bf16* lo, const __bf16* hi) {
  v8bf a = *(const v8bf*)lo;
  v8bf b = *(const v8bf*)hi;
  v16bf r;
#pragma unroll
  for (int i = 0; i < 8; ++i) { r[i] = a[i]; r[i + 8] = b[i]; }
  return r;
}

// Async stage of one 16-byte packet into LDS (CDNA5 async-to-LDS path, ASYNCcnt).
__device__ __forceinline__ void async_cp16(const __bf16* gsrc, __bf16* ldst) {
  unsigned lds_off = (unsigned)(uintptr_t)ldst;           // addr[31:0] = LDS offset
  unsigned long long ga = (unsigned long long)(uintptr_t)gsrc;
  asm volatile("global_load_async_to_lds_b128 %0, %1, off"
               :: "v"(lds_off), "v"(ga) : "memory");
}

// ---------------- f32 -> bf16 streaming convert ----------------
__global__ void cvt_f32_bf16(const float* __restrict__ x, __bf16* __restrict__ y, size_t n) {
  size_t i = (size_t)blockIdx.x * blockDim.x + threadIdx.x;
  if (i < n) y[i] = (__bf16)x[i];
}

// ---------------- LDS-tiled WMMA GEMM: C[M,N] = A[M,K] * W[N,K]^T ----------------
// 256 threads / 8 waves per block. Block tile 128x128; waves 2(M) x 4(N); each wave
// owns 8 16x16 WMMA tiles. K swept in KC=64 chunks through double-buffered LDS,
// staged with global_load_async_to_lds_b128 and drained with s_wait_asynccnt.
__global__ __launch_bounds__(256) void gemm_bf16_wmma_lds(const __bf16* __restrict__ A,
                                                          const __bf16* __restrict__ W,
                                                          float* __restrict__ C,
                                                          int M, int N, int K) {
  __shared__ __bf16 At[2][128][KC];   // 16 KB x2
  __shared__ __bf16 Bt[2][128][KC];   // 16 KB x2
  const int tid  = threadIdx.x;
  const int lane = tid & 31;
  const int wave = tid >> 5;
  const int wm   = wave & 1;          // 2 M-groups of 64 rows
  const int wn   = wave >> 1;         // 4 N-groups of 32 cols
  const int half = lane >> 4;
  const int rc   = lane & 15;
  const int nb   = N >> 7;
  const int bm   = blockIdx.x / nb;
  const int bn   = blockIdx.x - bm * nb;
  const int m0   = bm << 7;
  const int n0   = bn << 7;

  v8f acc[8];
#pragma unroll
  for (int t = 0; t < 8; ++t) acc[t] = vzero8();

  // each thread stages 4x16B for A and 4x16B for B per chunk (8 async ops)
  auto stage = [&](int buf, int c) {
    const size_t kk = (size_t)c * KC;
#pragma unroll
    for (int t = 0; t < 4; ++t) {
      const int off = (tid + 256 * t) * 8;       // element offset into 128xKC tile
      const int row = off >> 6;                  // /KC
      const int col = off & (KC - 1);
      async_cp16(A + (size_t)(m0 + row) * K + kk + col, &At[buf][row][col]);
      async_cp16(W + (size_t)(n0 + row) * K + kk + col, &Bt[buf][row][col]);
    }
  };

  const int NC = K / KC;
  stage(0, 0);
  for (int c = 0; c < NC; ++c) {
    const int buf = c & 1;
    if (c + 1 < NC) {
      stage(buf ^ 1, c + 1);
      asm volatile("s_wait_asynccnt 0x8" ::: "memory");  // chunk c's 8 ops done
    } else {
      asm volatile("s_wait_asynccnt 0x0" ::: "memory");
    }
    __syncthreads();                                      // all waves staged chunk c
#pragma unroll
    for (int ks = 0; ks < 2; ++ks) {                      // two 32-wide K steps
      v16bf bfr[2];
#pragma unroll
      for (int bb = 0; bb < 2; ++bb) {
        const __bf16* bp = &Bt[buf][wn * 32 + bb * 16 + rc][ks * 32 + half * 16];
        bfr[bb] = pack16(bp, bp + 8);
      }
#pragma unroll
      for (int aa = 0; aa < 4; ++aa) {
        const __bf16* ap = &At[buf][wm * 64 + aa * 16 + rc][ks * 32 + half * 16];
        v16bf afr = pack16(ap, ap + 8);
#pragma unroll
        for (int bb = 0; bb < 2; ++bb)
          acc[aa * 2 + bb] = __builtin_amdgcn_wmma_f32_16x16x32_bf16(
              false, afr, false, bfr[bb], (short)0, acc[aa * 2 + bb], false, false);
      }
    }
    __syncthreads();                                      // done reading buf
  }

#pragma unroll
  for (int aa = 0; aa < 4; ++aa) {
#pragma unroll
    for (int bb = 0; bb < 2; ++bb) {
      float* crow = C + (size_t)(m0 + wm * 64 + aa * 16 + 8 * half) * N
                      + n0 + wn * 32 + bb * 16 + rc;
#pragma unroll
      for (int r = 0; r < 8; ++r) crow[(size_t)r * N] = acc[aa * 2 + bb][r];
    }
  }
}

// ---------------- fused RMS-norm + RoPE, f32 in -> bf16 head-major out ----------
__global__ __launch_bounds__(256) void rmsnorm_rope(const float* __restrict__ x,
                                                    const float* __restrict__ w,
                                                    const float* __restrict__ cosb,
                                                    const float* __restrict__ sinb,
                                                    __bf16* __restrict__ out, int nh) {
  __shared__ float red[DH];
  __shared__ float nv[DH];
  const int bid = blockIdx.x;
  const int h = bid % nh;
  const int m = bid / nh;            // m = b*S + s
  const int s = m & (S_ - 1);
  const int b = m / S_;
  const int d = threadIdx.x;
  float v = x[(size_t)m * ((size_t)nh * DH) + (size_t)h * DH + d];
  red[d] = v * v;
  __syncthreads();
#pragma unroll
  for (int off = DH / 2; off > 0; off >>= 1) {
    if (d < off) red[d] += red[d + off];
    __syncthreads();
  }
  const float inv = rsqrtf(red[0] * (1.0f / (float)DH) + 1e-6f);
  const float n = v * inv * (1.0f + w[d]);
  nv[d] = n;
  __syncthreads();
  const float part = nv[d ^ (DH / 2)];
  const float rot = (d < DH / 2) ? -part : part;
  const float c  = cosb[(size_t)m * DH + d];
  const float sn = sinb[(size_t)m * DH + d];
  const float o = n * c + rot * sn;
  out[(((size_t)(b * nh + h) * S_) + s) * DH + d] = (__bf16)o;
}

// ---------------- V f32 [B*S, NKV*DH] -> bf16 transposed [B, NKV, DH, S] -------
__global__ void v_transpose(const float* __restrict__ v, __bf16* __restrict__ vt) {
  size_t i = (size_t)blockIdx.x * blockDim.x + threadIdx.x;
  const int col = (int)(i & (NKV * DH - 1));
  const size_t m = i >> 10;                                  // NKV*DH = 1024
  const int s = (int)(m & (S_ - 1));
  const int b = (int)(m / S_);
  const int kv = col >> 8;
  const int d  = col & (DH - 1);
  vt[(((size_t)(b * NKV + kv) * DH + d) * S_) + s] = (__bf16)v[i];
}

// ---------------- sliding-window flash attention (WMMA bf16) -------------------
__global__ __launch_bounds__(256) void attn_fa(const __bf16* __restrict__ q,
                                               const __bf16* __restrict__ k,
                                               const __bf16* __restrict__ vt,
                                               __bf16* __restrict__ o) {
  const int blk  = blockIdx.x;
  const int qblk = blk & 15;
  const int t2   = blk >> 4;
  const int h    = t2 & (NH - 1);
  const int b    = t2 >> 3;
  const int kv   = h >> 1;              // n_rep = 2
  const int wave = threadIdx.x >> 5;
  const int lane = threadIdx.x & 31;
  const int half = lane >> 4;
  const int rc   = lane & 15;
  const int q0   = qblk * 128 + wave * 16;

  const __bf16* qrow  = q  + ((size_t)(b * NH + h) * S_ + q0 + rc) * DH;
  const __bf16* kbase = k  + (size_t)(b * NKV + kv) * S_ * DH;
  const __bf16* vbase = vt + (size_t)(b * NKV + kv) * DH * (size_t)S_;

  __shared__ __bf16 plds[8][16 * 32];
  __bf16* myp = plds[wave];

  v8f acc[16];
#pragma unroll
  for (int t = 0; t < 16; ++t) acc[t] = vzero8();
  float mx[8], ls[8];
#pragma unroll
  for (int r = 0; r < 8; ++r) { mx[r] = -1e30f; ls[r] = 0.0f; }

  const int kb_start = (q0 > (WINDOW - 1)) ? ((q0 - (WINDOW - 1)) & ~31) : 0;
  const int kb_end   = q0 + 16;

  for (int kbp = kb_start; kbp < kb_end; kbp += 32) {
    v8f s0 = vzero8();
    v8f s1 = vzero8();
#pragma unroll
    for (int c = 0; c < 8; ++c) {
      v16bf qf = pack16(qrow + c * 32 + half * 16, qrow + c * 32 + half * 16 + 8);
      const __bf16* k0 = kbase + (size_t)(kbp + rc) * DH + c * 32 + half * 16;
      v16bf b0 = pack16(k0, k0 + 8);
      s0 = __builtin_amdgcn_wmma_f32_16x16x32_bf16(false, qf, false, b0,
                                                   (short)0, s0, false, false);
      const __bf16* k1 = k0 + 16 * DH;
      v16bf b1 = pack16(k1, k1 + 8);
      s1 = __builtin_amdgcn_wmma_f32_16x16x32_bf16(false, qf, false, b1,
                                                   (short)0, s1, false, false);
    }
    float alpha[8];
#pragma unroll
    for (int r = 0; r < 8; ++r) {
      const int i  = q0 + r + 8 * half;
      const int j0 = kbp + rc;
      const int j1 = kbp + 16 + rc;
      float x0 = SOFTCAP * tanhf(s0[r] * (SCALE / SOFTCAP));
      float x1 = SOFTCAP * tanhf(s1[r] * (SCALE / SOFTCAP));
      if (!((j0 <= i) && (j0 > i - WINDOW))) x0 = -1e9f;
      if (!((j1 <= i) && (j1 > i - WINDOW))) x1 = -1e9f;
      float rmax = fmaxf(x0, x1);
#pragma unroll
      for (int ofs = 8; ofs >= 1; ofs >>= 1)
        rmax = fmaxf(rmax, __shfl_xor(rmax, ofs, 32));
      const float mnew = fmaxf(mx[r], rmax);
      const float a = __expf(mx[r] - mnew);
      alpha[r] = a;
      const float p0 = __expf(x0 - mnew);
      const float p1 = __expf(x1 - mnew);
      float ps = p0 + p1;
#pragma unroll
      for (int ofs = 8; ofs >= 1; ofs >>= 1)
        ps += __shfl_xor(ps, ofs, 32);
      ls[r] = ls[r] * a + ps;
      mx[r] = mnew;
      const int mrow = r + 8 * half;
      myp[mrow * 32 + rc]      = (__bf16)p0;
      myp[mrow * 32 + 16 + rc] = (__bf16)p1;
    }
    asm volatile("s_wait_dscnt 0" ::: "memory");
    v16bf pf = pack16(myp + rc * 32 + half * 16, myp + rc * 32 + half * 16 + 8);
#pragma unroll
    for (int t = 0; t < 16; ++t) {
#pragma unroll
      for (int r = 0; r < 8; ++r) acc[t][r] *= alpha[r];
    }
#pragma unroll
    for (int t = 0; t < 16; ++t) {
      const __bf16* vr = vbase + (size_t)(t * 16 + rc) * S_ + kbp + half * 16;
      v16bf vfr = pack16(vr, vr + 8);
      acc[t] = __builtin_amdgcn_wmma_f32_16x16x32_bf16(false, pf, false, vfr,
                                                       (short)0, acc[t], false, false);
    }
  }
#pragma unroll
  for (int t = 0; t < 16; ++t) {
#pragma unroll
    for (int r = 0; r < 8; ++r) {
      const int i = q0 + r + 8 * half;
      const float val = acc[t][r] / ls[r];
      o[(size_t)(b * S_ + i) * ((size_t)NH * DH) + h * DH + t * 16 + rc] = (__bf16)val;
    }
  }
}

extern "C" void kernel_launch(void* const* d_in, const int* in_sizes, int n_in,
                              void* d_out, int out_size, void* d_ws, size_t ws_size,
                              hipStream_t stream) {
  (void)in_sizes; (void)n_in; (void)out_size; (void)ws_size;
  const float* hs   = (const float*)d_in[0];
  const float* cosb = (const float*)d_in[1];
  const float* sinb = (const float*)d_in[2];
  /* d_in[3] = attention_mask: recomputed analytically on-device */
  const float* wq   = (const float*)d_in[4];
  const float* wk   = (const float*)d_in[5];
  const float* wv   = (const float*)d_in[6];
  const float* wo   = (const float*)d_in[7];
  const float* qnw  = (const float*)d_in[8];
  const float* knw  = (const float*)d_in[9];
  float* out = (float*)d_out;

  const size_t M   = (size_t)B_ * S_;          // 4096
  const size_t nHS = M * H_;
  const size_t nWQ = (size_t)NH * DH * H_;
  const size_t nWK = (size_t)NKV * DH * H_;
  const size_t nWO = (size_t)H_ * NH * DH;
  const size_t nQ  = M * NH * DH;
  const size_t nK  = M * NKV * DH;

  char* p = (char*)d_ws;
  size_t off = 0;
  auto take = [&](size_t bytes) -> char* {
    char* r = p + off;
    off += (bytes + 255) & ~(size_t)255;
    return r;
  };
  __bf16* hsb  = (__bf16*)take(nHS * 2);
  __bf16* wqb  = (__bf16*)take(nWQ * 2);
  __bf16* wkb  = (__bf16*)take(nWK * 2);
  __bf16* wvb  = (__bf16*)take(nWK * 2);
  __bf16* wob  = (__bf16*)take(nWO * 2);
  float*  qf   = (float*)take(nQ * 4);
  float*  kf   = (float*)take(nK * 4);
  float*  vf   = (float*)take(nK * 4);
  __bf16* qb   = (__bf16*)take(nQ * 2);
  __bf16* kb16 = (__bf16*)take(nK * 2);
  __bf16* attb = (__bf16*)qf;   // alias: qf dead after rope(q)
  __bf16* vtb  = (__bf16*)kf;   // alias: kf dead after rope(k)

  // 1) bf16 conversions
  cvt_f32_bf16<<<(unsigned)((nHS + 255) / 256), 256, 0, stream>>>(hs, hsb, nHS);
  cvt_f32_bf16<<<(unsigned)((nWQ + 255) / 256), 256, 0, stream>>>(wq, wqb, nWQ);
  cvt_f32_bf16<<<(unsigned)((nWK + 255) / 256), 256, 0, stream>>>(wk, wkb, nWK);
  cvt_f32_bf16<<<(unsigned)((nWK + 255) / 256), 256, 0, stream>>>(wv, wvb, nWK);
  cvt_f32_bf16<<<(unsigned)((nWO + 255) / 256), 256, 0, stream>>>(wo, wob, nWO);

  // 2) QKV projections: grid = (M/128)*(N/128)
  gemm_bf16_wmma_lds<<<32 * 16, 256, 0, stream>>>(hsb, wqb, qf, (int)M, NH * DH, H_);
  gemm_bf16_wmma_lds<<<32 * 8,  256, 0, stream>>>(hsb, wkb, kf, (int)M, NKV * DH, H_);
  gemm_bf16_wmma_lds<<<32 * 8,  256, 0, stream>>>(hsb, wvb, vf, (int)M, NKV * DH, H_);

  // 3) RMS-norm + RoPE (q,k) and V transpose, all to bf16
  rmsnorm_rope<<<(unsigned)(M * NH),  256, 0, stream>>>(qf, qnw, cosb, sinb, qb,  NH);
  rmsnorm_rope<<<(unsigned)(M * NKV), 256, 0, stream>>>(kf, knw, cosb, sinb, kb16, NKV);
  v_transpose<<<(unsigned)(nK / 256), 256, 0, stream>>>(vf, vtb);

  // 4) sliding-window flash attention
  attn_fa<<<B_ * NH * (S_ / 128), 256, 0, stream>>>(qb, kb16, vtb, attb);

  // 5) output projection -> f32 d_out
  gemm_bf16_wmma_lds<<<32 * 20, 256, 0, stream>>>(attb, wob, out, (int)M, H_, NH * DH);
}
